// PC_RNN_HC_A_14413910245618
// MI455X (gfx1250) — matrix operand order repaired and checked by reading, not compiled, not run on hardware
//
#include <hip/hip_runtime.h>
#include <math.h>
#include <stddef.h>
#include <stdint.h>

// ---------------- problem constants (from reference) ----------------
#define SEQ_LEN 512
#define BATCH   256
#define CAUSES  64
#define STATES  512
#define OUTD    256
#define INV_TAU 0.1f
#define ALPHA_X 0.1f
#define ALPHA_H 0.1f

// ---------------- launch configuration ----------------
#define NWG 64          // persistent workgroups (one B-column set each, 58KB LDS)
#define TPB 256         // 8 wave32 per workgroup -> 512 waves total
#define WPG (TPB / 32)

typedef __attribute__((ext_vector_type(16))) __bf16 v16bf;
typedef __attribute__((ext_vector_type(8)))  __bf16 v8bf;
typedef __attribute__((ext_vector_type(8)))  float  v8f;

#define WMMA_BF16(a, b, c) \
  __builtin_amdgcn_wmma_f32_16x16x32_bf16(false, (a), false, (b), (short)0, (c), false, false)

// ---- hardware tanh (CDNA5 v_tanh_f32); branch-free fallback via v_exp/v_rcp ----
#if defined(__has_builtin)
#if __has_builtin(__builtin_amdgcn_tanhf)
#define TANHF(xx) __builtin_amdgcn_tanhf(xx)
#endif
#endif
#ifndef TANHF
__device__ __forceinline__ float fast_tanh_impl(float x) {
  const float e2 = __builtin_amdgcn_exp2f(x * 2.8853900817779268f);
  return 1.0f - 2.0f * __builtin_amdgcn_rcpf(e2 + 1.0f);
}
#define TANHF(xx) fast_tanh_impl(xx)
#endif

struct WS {
  __bf16*   tpost;   // bf16(tanh(h_post))   (BATCH, STATES)  WMMA A operand
  __bf16*   tprior;  // bf16(tanh(h_prior))  (BATCH, STATES)  WMMA A operand
  __bf16*   errb;    // bf16(error)          (BATCH, OUTD)    WMMA A operand
  __bf16*   ehb;     // bf16(error_h)        (BATCH, STATES)  WMMA A operand
  __bf16*   cbf;     // bf16(c)              (BATCH, CAUSES)  WMMA A operand
  __bf16*   wrT;     // packed B: w_r.T   K=STATES, N=STATES
  __bf16*   wcT;     // packed B: w_c.T   K=CAUSES, N=STATES
  __bf16*   woT;     // packed B: w_o.T   K=STATES, N=OUTD
  __bf16*   wo;      // packed B: w_o     K=OUTD,   N=STATES
  __bf16*   wc;      // packed B: w_c     K=STATES, N=CAUSES
  unsigned* bar;     // [0]=count, [1]=generation
};

// -------- A operand: 16x32 bf16 tile from row-major activation matrix --------
// ISA 16-bit A layout: lanes 0-15 hold row M=lane, K in {0..7, 16..23};
// lanes 16-31 hold row M=lane-16, K in {8..15, 24..31}.
__device__ __forceinline__ v16bf load_a(const __bf16* __restrict__ A, int ldA,
                                        int m0, int k0, int lane) {
  const int row  = m0 + (lane & 15);
  const int half = (lane >> 4) & 1;
  const __bf16* base = A + (size_t)row * ldA + k0;
  union { struct { v8bf lo; v8bf hi; } s; v16bf v; } u;
  u.s.lo = *(const v8bf*)(base + half * 8);        // K = k0 + half*8 + [0..7]
  u.s.hi = *(const v8bf*)(base + 16 + half * 8);   // K = k0 + 16 + half*8 + [0..7]
  return u.v;
}

// -------- B operand from LDS column (nt fixed per block): ds_load_b128 x2 -----
__device__ __forceinline__ v16bf load_b_lds(const __bf16* L, int kt, int lane) {
  return *(const v16bf*)(L + (((kt << 5) + lane) << 4));
}

// -------- one-time async global->LDS stage of a packed B column ---------------
// CDNA5 GLOBAL_LOAD_ASYNC_TO_LDS_B128: each lane copies 16B global->LDS, counted by ASYNCcnt.
__device__ __forceinline__ void async_stage(const __bf16* __restrict__ gsrc,
                                            __bf16* lds_dst, int n_elems) {
  const uint32_t lbase = (uint32_t)(uintptr_t)(void*)lds_dst;  // low 32b = LDS offset
  for (int i = threadIdx.x; (i << 3) < n_elems; i += TPB) {
    const uint32_t laddr = lbase + (uint32_t)(i << 4);
    const unsigned long long gaddr = (unsigned long long)(uintptr_t)(gsrc + (i << 3));
    asm volatile("global_load_async_to_lds_b128 %0, %1, off"
                 :: "v"(laddr), "v"(gaddr) : "memory");
  }
}

// -------- device-scope sense barrier (L2 atomics), counters self-reset --------
__device__ __forceinline__ void grid_sync(unsigned* bar) {
  __syncthreads();
  if (threadIdx.x == 0) {
    __threadfence();
    unsigned gen = __hip_atomic_load(&bar[1], __ATOMIC_RELAXED, __HIP_MEMORY_SCOPE_AGENT);
    unsigned n = __hip_atomic_fetch_add(&bar[0], 1u, __ATOMIC_ACQ_REL, __HIP_MEMORY_SCOPE_AGENT);
    if (n == NWG - 1) {
      __hip_atomic_store(&bar[0], 0u, __ATOMIC_RELAXED, __HIP_MEMORY_SCOPE_AGENT);
      __hip_atomic_fetch_add(&bar[1], 1u, __ATOMIC_RELEASE, __HIP_MEMORY_SCOPE_AGENT);
    } else {
      while (__hip_atomic_load(&bar[1], __ATOMIC_ACQUIRE, __HIP_MEMORY_SCOPE_AGENT) == gen) {
        __builtin_amdgcn_s_sleep(2);
      }
    }
  }
  __syncthreads();
}

// ---------------- prep: pack a weight matrix into WMMA-B tile layout ----------
// Logical B is (K, N). transposed=1 -> B[k][n] = src[n*K + k], else src[k*N + n].
__global__ void pack_weight(const float* __restrict__ src, __bf16* __restrict__ dst,
                            int K, int N, int transposed) {
  const int total = (K >> 5) * (N >> 4) * 512;
  const int ktiles = K >> 5;
  for (int p = blockIdx.x * blockDim.x + threadIdx.x; p < total;
       p += gridDim.x * blockDim.x) {
    const int e = p & 15;
    const int lane = (p >> 4) & 31;
    const int tile = p >> 9;
    const int kt = tile % ktiles;
    const int nt = tile / ktiles;
    const int k = (kt << 5) + ((lane >> 4) & 1) * 8 + (e & 7) + ((e >> 3) << 4);
    const int n = (nt << 4) + (lane & 15);
    const float v = transposed ? src[(size_t)n * K + k] : src[(size_t)k * N + n];
    dst[p] = (__bf16)v;
  }
}

// ---------------- prep: initialize bf16 operand copies + barrier ----------------
__global__ void init_state(WS ws, const float* __restrict__ c_init,
                           const float* __restrict__ h_init) {
  const int tid = blockIdx.x * blockDim.x + threadIdx.x;
  const int stride = gridDim.x * blockDim.x;
  if (tid == 0) { ws.bar[0] = 0u; ws.bar[1] = 0u; }
  for (int i = tid; i < BATCH * STATES; i += stride)
    ws.tpost[i] = (__bf16)TANHF(h_init[i]);
  for (int i = tid; i < BATCH * CAUSES; i += stride)
    ws.cbf[i] = (__bf16)c_init[i];
}

// ---------------- persistent cooperative scan kernel ----------------
// Recurrent state (h_post, h_prior, tanh(h_prior), c) for this wave's fixed tile
// lives in VGPRs across all 512 steps; only bf16 operand copies hit memory.
__global__ void __launch_bounds__(TPB)
pcrnn_persistent(WS ws, const float* __restrict__ x, const float* __restrict__ b_o,
                 const float* __restrict__ b_r, const float* __restrict__ h_init,
                 const float* __restrict__ c_init, float* __restrict__ out) {
  // Per-block B columns resident in LDS for the whole sequence (58KB < 320KB WGP).
  __shared__ __attribute__((aligned(32))) __bf16 sWrT[16 * 512];  // 16KB
  __shared__ __attribute__((aligned(32))) __bf16 sWcT[ 2 * 512];  //  2KB
  __shared__ __attribute__((aligned(32))) __bf16 sWoT[16 * 512];  // 16KB
  __shared__ __attribute__((aligned(32))) __bf16 sWo [ 8 * 512];  //  8KB
  __shared__ __attribute__((aligned(32))) __bf16 sWc [16 * 512];  // 16KB

  const int lane  = threadIdx.x & 31;
  const int gw    = blockIdx.x * WPG + (threadIdx.x >> 5);  // global wave 0..511
  const int half  = lane >> 4;
  const int ncol  = lane & 15;
  const int mrow0 = half << 3;      // C/D: VGPR v -> row m0 + mrow0 + v
  const int mt    = gw & 15;        // row-tile: identical mapping in all phases
  const int nt    = blockIdx.x >> 1;// col-tile: shared by all 8 waves of a block
  const int m0    = mt << 4;
  const int n     = (nt << 4) + ncol;

  // ---- one-time async stage of this block's B columns into LDS ----
  async_stage(ws.wrT + (size_t)nt * (16 * 512), sWrT, 16 * 512);
  async_stage(ws.wcT + (size_t)nt * ( 2 * 512), sWcT,  2 * 512);
  async_stage(ws.wo  + (size_t)nt * ( 8 * 512), sWo,   8 * 512);
  if (blockIdx.x < 32) async_stage(ws.woT + (size_t)nt * (16 * 512), sWoT, 16 * 512);
  if (blockIdx.x < 8)  async_stage(ws.wc  + (size_t)nt * (16 * 512), sWc,  16 * 512);
  asm volatile("s_wait_asynccnt 0x0" ::: "memory");
  __syncthreads();

  // loop-invariant biases for this wave's output columns
  const float br = b_r[n];
  const float bo = (blockIdx.x < 32) ? b_o[n] : 0.0f;

  // ---- register-resident recurrent state for this wave's tile ----
  float hreg[8];   // h_post rows m0+mrow0+v, column n (STATES space)
  float hpreg[8];  // h_prior (phase1 -> phase3, same wave)
  float tpreg[8];  // tanh(h_prior) in f32 (phase1 -> phase3, same wave)
  float creg[8];   // c rows, column n (CAUSES space), blocks 0..7 only
#pragma unroll
  for (int v = 0; v < 8; ++v) {
    hreg[v] = h_init[(size_t)(m0 + mrow0 + v) * STATES + n];
    hpreg[v] = 0.0f;
    tpreg[v] = 0.0f;
  }
#pragma unroll
  for (int v = 0; v < 8; ++v)
    creg[v] = (blockIdx.x < 8) ? c_init[(size_t)(m0 + mrow0 + v) * CAUSES + n] : 0.0f;

  for (int t = 0; t < SEQ_LEN; ++t) {
    // ---- Phase 1: h_prior = 0.9*h + 0.1*(tanh(h)@WrT + c@WcT + b_r); tprior ----
    {
      v8f acc = {};
#pragma unroll 4
      for (int kt = 0; kt < STATES / 32; ++kt) {
        v16bf a = load_a(ws.tpost, STATES, m0, kt << 5, lane);
        v16bf b = load_b_lds(sWrT, kt, lane);
        acc = WMMA_BF16(a, b, acc);
      }
#pragma unroll
      for (int kt = 0; kt < CAUSES / 32; ++kt) {
        v16bf a = load_a(ws.cbf, CAUSES, m0, kt << 5, lane);
        v16bf b = load_b_lds(sWcT, kt, lane);
        acc = WMMA_BF16(a, b, acc);
      }
#pragma unroll
      for (int v = 0; v < 8; ++v) {
        const float hpv = (1.0f - INV_TAU) * hreg[v] + INV_TAU * (acc[v] + br);
        const float tpv = TANHF(hpv);
        hpreg[v] = hpv;
        tpreg[v] = tpv;
        ws.tprior[(size_t)(m0 + mrow0 + v) * STATES + n] = (__bf16)tpv;
      }
    }
    grid_sync(ws.bar);

    // ---- Phase 2: error = tanh(h_prior)@WoT + b_o - x[t]  (blocks 0..31) ----
    if (blockIdx.x < 32) {
      v8f acc = {};
#pragma unroll 4
      for (int kt = 0; kt < STATES / 32; ++kt) {
        v16bf a = load_a(ws.tprior, STATES, m0, kt << 5, lane);
        v16bf b = load_b_lds(sWoT, kt, lane);
        acc = WMMA_BF16(a, b, acc);
      }
      const float* xt = x + (size_t)t * BATCH * OUTD;
      float* ot = out + (size_t)t * BATCH * OUTD;
#pragma unroll
      for (int v = 0; v < 8; ++v) {
        const size_t idx = (size_t)(m0 + mrow0 + v) * OUTD + n;
        const float e = acc[v] + bo - xt[idx];
        ot[idx] = e;
        ws.errb[idx] = (__bf16)e;
      }
      if (t + 1 < SEQ_LEN)  // global_prefetch_b8: warm next timestep of x
        __builtin_prefetch(x + (size_t)(t + 1) * BATCH * OUTD +
                               (size_t)(m0 + mrow0) * OUTD + n, 0, 1);
    }
    grid_sync(ws.bar);

    // ---- Phase 3: g = error@Wo; h_post = h_prior - ax*(1-t^2)*g; tpost; ehb ----
    {
      v8f acc = {};
#pragma unroll 4
      for (int kt = 0; kt < OUTD / 32; ++kt) {
        v16bf a = load_a(ws.errb, OUTD, m0, kt << 5, lane);
        v16bf b = load_b_lds(sWo, kt, lane);
        acc = WMMA_BF16(a, b, acc);
      }
#pragma unroll
      for (int v = 0; v < 8; ++v) {
        const size_t idx = (size_t)(m0 + mrow0 + v) * STATES + n;
        const float tp = tpreg[v];
        const float delta = ALPHA_X * (1.0f - tp * tp) * acc[v];  // = error_h
        const float hpost = hpreg[v] - delta;
        hreg[v] = hpost;
        ws.tpost[idx] = (__bf16)TANHF(hpost);
        ws.ehb[idx] = (__bf16)delta;
      }
    }
    grid_sync(ws.bar);

    // ---- Phase 4: c -= ah * (error_h @ Wc)   (blocks 0..7) ----
    if (blockIdx.x < 8) {
      v8f acc = {};
#pragma unroll 4
      for (int kt = 0; kt < STATES / 32; ++kt) {
        v16bf a = load_a(ws.ehb, STATES, m0, kt << 5, lane);
        v16bf b = load_b_lds(sWc, kt, lane);
        acc = WMMA_BF16(a, b, acc);
      }
#pragma unroll
      for (int v = 0; v < 8; ++v) {
        const float cn = creg[v] - ALPHA_H * acc[v];
        creg[v] = cn;
        ws.cbf[(size_t)(m0 + mrow0 + v) * CAUSES + n] = (__bf16)cn;
      }
    }
    grid_sync(ws.bar);
  }
}

extern "C" void kernel_launch(void* const* d_in, const int* in_sizes, int n_in,
                              void* d_out, int out_size, void* d_ws, size_t ws_size,
                              hipStream_t stream) {
  (void)in_sizes; (void)n_in; (void)out_size; (void)ws_size;
  const float* x      = (const float*)d_in[0];
  const float* c_init = (const float*)d_in[1];
  const float* h_init = (const float*)d_in[2];
  const float* w_o    = (const float*)d_in[3];
  const float* b_o    = (const float*)d_in[4];
  const float* w_c    = (const float*)d_in[5];
  const float* w_r    = (const float*)d_in[6];
  const float* b_r    = (const float*)d_in[7];

  char* base = (char*)d_ws;
  size_t off = 0;
  auto take = [&](size_t bytes) -> char* {
    char* p = base + off;
    off = (off + bytes + 255) & ~(size_t)255;
    return p;
  };

  WS ws;
  ws.bar    = (unsigned*)take(256);
  ws.tpost  = (__bf16*)take(2 * (size_t)BATCH * STATES);
  ws.tprior = (__bf16*)take(2 * (size_t)BATCH * STATES);
  ws.errb   = (__bf16*)take(2 * (size_t)BATCH * OUTD);
  ws.ehb    = (__bf16*)take(2 * (size_t)BATCH * STATES);
  ws.cbf    = (__bf16*)take(2 * (size_t)BATCH * CAUSES);
  ws.wrT    = (__bf16*)take(2 * (size_t)STATES * STATES);
  ws.wcT    = (__bf16*)take(2 * (size_t)CAUSES * STATES);
  ws.woT    = (__bf16*)take(2 * (size_t)STATES * OUTD);
  ws.wo     = (__bf16*)take(2 * (size_t)OUTD * STATES);
  ws.wc     = (__bf16*)take(2 * (size_t)STATES * CAUSES);

  // pack weights into WMMA B-operand layout (transpose folded into the pack)
  pack_weight<<<128, 256, 0, stream>>>(w_r, ws.wrT, STATES, STATES, 1); // w_r.T
  pack_weight<<<32,  256, 0, stream>>>(w_c, ws.wcT, CAUSES, STATES, 1); // w_c.T
  pack_weight<<<64,  256, 0, stream>>>(w_o, ws.woT, STATES, OUTD,   1); // w_o.T
  pack_weight<<<64,  256, 0, stream>>>(w_o, ws.wo,  OUTD,   STATES, 0); // w_o
  pack_weight<<<32,  256, 0, stream>>>(w_c, ws.wc,  STATES, CAUSES, 0); // w_c

  init_state<<<128, 256, 0, stream>>>(ws, c_init, h_init);

  pcrnn_persistent<<<NWG, TPB, 0, stream>>>(ws, x, b_o, b_r, h_init, c_init,
                                            (float*)d_out);
}
